// KGATLayer_38611755991248
// MI455X (gfx1250) — compile-verified, matrix operand order checked by reading, not run on hardware
//
#include <hip/hip_runtime.h>

typedef __attribute__((ext_vector_type(16))) _Float16 v16h;
typedef __attribute__((ext_vector_type(8)))  float    v8f;

#define DMODEL 128
#define NHEADS 4
#define HD 32

// ---------------- monotone float<->uint order map (for global atomic max) ----
__device__ __forceinline__ unsigned fmap(float x) {
    unsigned b = __float_as_uint(x);
    return (b & 0x80000000u) ? ~b : (b | 0x80000000u);
}
__device__ __forceinline__ float funmap(unsigned u) {
    unsigned b = (u & 0x80000000u) ? (u ^ 0x80000000u) : ~u;
    return __uint_as_float(b);
}

// ---------------- WMMA GEMM: Y[n,o] = sum_k X[n,k] * W[o,k]  (x @ W.T) -------
// Block = 128 threads = 4 waves. Each wave computes a 16(M) x 128(N) stripe.
// W (128x128 f32) staged once to LDS as f16 [o][k]; per-wave X tile staged as
// f16 [16][128]. K=128 consumed as 4 chained v_wmma_f32_16x16x32_f16.
__global__ __launch_bounds__(128) void proj_wmma(const float* __restrict__ X,
                                                 const float* __restrict__ Wm,
                                                 float* __restrict__ Y,
                                                 int nrows) {
    __shared__ _Float16 Wl[DMODEL * DMODEL];     // 32 KB
    __shared__ _Float16 Xl[4][16 * DMODEL];      // 16 KB
    const int tid  = threadIdx.x;
    const int wave = tid >> 5;
    const int lane = tid & 31;

    // stage W -> LDS f16 (cooperative, vectorized)
    for (int i = tid * 4; i < DMODEL * DMODEL; i += 128 * 4) {
        float4 v = *(const float4*)(Wm + i);
        Wl[i + 0] = (_Float16)v.x; Wl[i + 1] = (_Float16)v.y;
        Wl[i + 2] = (_Float16)v.z; Wl[i + 3] = (_Float16)v.w;
    }
    // stage this wave's 16-row X tile -> LDS f16
    const int row0 = blockIdx.x * 64 + wave * 16;
    for (int i = lane * 4; i < 16 * DMODEL; i += 32 * 4) {
        int rr = i >> 7, cc = i & (DMODEL - 1);
        int gr = row0 + rr;
        float4 v = (gr < nrows) ? *(const float4*)(X + (size_t)gr * DMODEL + cc)
                                : make_float4(0.f, 0.f, 0.f, 0.f);
        Xl[wave][i + 0] = (_Float16)v.x; Xl[wave][i + 1] = (_Float16)v.y;
        Xl[wave][i + 2] = (_Float16)v.z; Xl[wave][i + 3] = (_Float16)v.w;
    }
    __syncthreads();

    const int kbase = (lane >= 16) ? 8 : 0;   // ISA 16-bit A/B fragment K base
    const int mloc  = lane & 15;

    v8f acc[8];
#pragma unroll
    for (int nt = 0; nt < 8; ++nt)
#pragma unroll
        for (int i = 0; i < 8; ++i) acc[nt][i] = 0.0f;

#pragma unroll
    for (int kc = 0; kc < 4; ++kc) {
        // A fragment: rows of X tile
        v16h a;
        const _Float16* ab = &Xl[wave][mloc * DMODEL + kc * 32];
#pragma unroll
        for (int e = 0; e < 8; ++e) { a[e] = ab[kbase + e]; a[8 + e] = ab[16 + kbase + e]; }
#pragma unroll
        for (int nt = 0; nt < 8; ++nt) {
            // B fragment: B[k, n] = W[n_global, k]; column n held per lane
            v16h b;
            const _Float16* bb = &Wl[(nt * 16 + mloc) * DMODEL + kc * 32];
#pragma unroll
            for (int e = 0; e < 8; ++e) { b[e] = bb[kbase + e]; b[8 + e] = bb[16 + kbase + e]; }
            acc[nt] = __builtin_amdgcn_wmma_f32_16x16x32_f16(
                false, a, false, b, (short)0, acc[nt], false, false);
        }
    }

    // D layout: VGPR r -> M = r (lanes 0-15) or 8+r (lanes 16-31); N = lane&15
    const int rbase = row0 + ((lane >= 16) ? 8 : 0);
    const int cbase = lane & 15;
#pragma unroll
    for (int nt = 0; nt < 8; ++nt)
#pragma unroll
        for (int r = 0; r < 8; ++r) {
            int row = rbase + r;
            if (row < nrows) Y[(size_t)row * DMODEL + nt * 16 + cbase] = acc[nt][r];
        }
}

// ---------------- init: zero out + attn_sum, reset max scratch ---------------
__global__ __launch_bounds__(256) void init_buffers(float* __restrict__ out,
                                                    float* __restrict__ attn_sum,
                                                    unsigned* __restrict__ maxscratch,
                                                    int nout, int nsum) {
    int idx = blockIdx.x * blockDim.x + threadIdx.x;
    int stride = gridDim.x * blockDim.x;
    for (int i = idx; i < nout; i += stride) out[i] = 0.0f;
    for (int i = idx; i < nsum; i += stride) attn_sum[i] = 0.0f;
    if (idx == 0) *maxscratch = 0u;   // below fmap(-inf): lower bound
}

// ---------------- per-node partial scores s1,s3 ------------------------------
__global__ __launch_bounds__(256) void node_scores(const float* __restrict__ h,
                                                   const float* __restrict__ aw,
                                                   float* __restrict__ s1,
                                                   float* __restrict__ s3, int nN) {
    int idx = blockIdx.x * blockDim.x + threadIdx.x;
    if (idx >= nN * NHEADS) return;
    int n = idx >> 2, hh = idx & 3;
    const float* p = h + (size_t)n * DMODEL + hh * HD;
    float d1 = 0.f, d3 = 0.f;
#pragma unroll
    for (int i = 0; i < HD; ++i) { float v = p[i]; d1 += v * aw[i]; d3 += v * aw[64 + i]; }
    s1[idx] = d1; s3[idx] = d3;
}

// ---------------- per-relation partial score s2 ------------------------------
__global__ __launch_bounds__(256) void rel_scores(const float* __restrict__ r,
                                                  const float* __restrict__ aw,
                                                  float* __restrict__ s2, int nR) {
    int idx = blockIdx.x * blockDim.x + threadIdx.x;
    if (idx >= nR * NHEADS) return;
    int n = idx >> 2, hh = idx & 3;
    const float* p = r + (size_t)n * DMODEL + hh * HD;
    float d2 = 0.f;
#pragma unroll
    for (int i = 0; i < HD; ++i) d2 += p[i] * aw[32 + i];
    s2[idx] = d2;
}

// ---------------- per-edge raw score + global max ----------------------------
__global__ __launch_bounds__(256) void edge_scores(const int* __restrict__ src,
                                                   const int* __restrict__ dst,
                                                   const int* __restrict__ etype,
                                                   const float* __restrict__ s1,
                                                   const float* __restrict__ s2,
                                                   const float* __restrict__ s3,
                                                   const float* __restrict__ attn_b,
                                                   float* __restrict__ raw,
                                                   unsigned* __restrict__ maxscratch,
                                                   int nE) {
    int idx = blockIdx.x * blockDim.x + threadIdx.x;
    float sc = -3.4e38f;
    if (idx < nE * NHEADS) {
        int e = idx >> 2, hh = idx & 3;
        float v = s1[src[e] * NHEADS + hh] + s2[etype[e] * NHEADS + hh] +
                  s3[dst[e] * NHEADS + hh] + attn_b[0];
        sc = (v >= 0.f) ? v : 0.2f * v;    // leaky_relu(0.2)
        raw[idx] = sc;
    }
    __shared__ float red[256];
    red[threadIdx.x] = sc;
    __syncthreads();
    for (int s = 128; s > 0; s >>= 1) {
        if (threadIdx.x < (unsigned)s)
            red[threadIdx.x] = fmaxf(red[threadIdx.x], red[threadIdx.x + s]);
        __syncthreads();
    }
    if (threadIdx.x == 0) atomicMax(maxscratch, fmap(red[0]));
}

// ---------------- exp(score - gmax) + segment-sum into attn_sum --------------
__global__ __launch_bounds__(256) void edge_exp_sum(const int* __restrict__ dst,
                                                    float* __restrict__ raw,
                                                    float* __restrict__ attn_sum,
                                                    const unsigned* __restrict__ maxscratch,
                                                    int nE) {
    int idx = blockIdx.x * blockDim.x + threadIdx.x;
    if (idx >= nE * NHEADS) return;
    float gmax = funmap(*maxscratch);
    int e = idx >> 2, hh = idx & 3;
    float se = expf(raw[idx] - gmax);
    raw[idx] = se;                                 // in-place: raw -> exp
    atomicAdd(&attn_sum[dst[e] * NHEADS + hh], se);
}

// ---------------- weighted scatter-add of messages ---------------------------
__global__ __launch_bounds__(256) void aggregate(const int* __restrict__ src,
                                                 const int* __restrict__ dst,
                                                 const float* __restrict__ h,
                                                 const float* __restrict__ sexp,
                                                 const float* __restrict__ attn_sum,
                                                 float* __restrict__ out, int total) {
    int idx = blockIdx.x * blockDim.x + threadIdx.x;
    int stride = gridDim.x * blockDim.x;
    for (int i = idx; i < total; i += stride) {
        int e = i >> 7;               // / 128
        int c = i & (DMODEL - 1);
        int hh = c >> 5;
        int s_ = src[e], d_ = dst[e];
        float w = sexp[e * NHEADS + hh] / (attn_sum[d_ * NHEADS + hh] + 1e-8f);
        atomicAdd(&out[(size_t)d_ * DMODEL + c], w * h[(size_t)s_ * DMODEL + c]);
    }
}

extern "C" void kernel_launch(void* const* d_in, const int* in_sizes, int n_in,
                              void* d_out, int out_size, void* d_ws, size_t ws_size,
                              hipStream_t stream) {
    const float* entity_emb = (const float*)d_in[0];
    const float* rel_emb    = (const float*)d_in[1];
    const int*   edge_index = (const int*)d_in[2];
    const int*   edge_type  = (const int*)d_in[3];
    const float* W          = (const float*)d_in[4];
    const float* W_r        = (const float*)d_in[5];
    const float* attn_w     = (const float*)d_in[6];
    const float* attn_b     = (const float*)d_in[7];
    float* out = (float*)d_out;

    const int N = in_sizes[0] / DMODEL;
    const int R = in_sizes[1] / DMODEL;
    const int E = in_sizes[3];
    const int* src = edge_index;
    const int* dst = edge_index + E;

    // ---- carve workspace ----
    char* base = (char*)d_ws;
    size_t off = 0;
    auto carve = [&](size_t bytes) -> char* {
        char* p = base + off;
        off = (off + bytes + 255) & ~(size_t)255;
        return p;
    };
    float*    h_ws     = (float*)carve((size_t)N * DMODEL * 4);
    float*    r_ws     = (float*)carve((size_t)R * DMODEL * 4);
    float*    s1       = (float*)carve((size_t)N * NHEADS * 4);
    float*    s3       = (float*)carve((size_t)N * NHEADS * 4);
    float*    s2       = (float*)carve((size_t)R * NHEADS * 4);
    float*    raw      = (float*)carve((size_t)E * NHEADS * 4);
    float*    attn_sum = (float*)carve((size_t)N * NHEADS * 4);
    unsigned* maxscr   = (unsigned*)carve(256);
    (void)ws_size; (void)n_in; (void)out_size;

    // 1) projections via WMMA
    proj_wmma<<<(N + 63) / 64, 128, 0, stream>>>(entity_emb, W, h_ws, N);
    proj_wmma<<<(R + 63) / 64, 128, 0, stream>>>(rel_emb, W_r, r_ws, R);

    // 2) zero out / attn_sum, reset global-max scratch
    init_buffers<<<2048, 256, 0, stream>>>(out, attn_sum, maxscr, N * DMODEL, N * NHEADS);

    // 3) separable attention partials
    node_scores<<<(N * NHEADS + 255) / 256, 256, 0, stream>>>(h_ws, attn_w, s1, s3, N);
    rel_scores<<<(R * NHEADS + 255) / 256, 256, 0, stream>>>(r_ws, attn_w, s2, R);

    // 4) per-edge scores + global max
    edge_scores<<<(E * NHEADS + 255) / 256, 256, 0, stream>>>(
        src, dst, edge_type, s1, s2, s3, attn_b, raw, maxscr, E);

    // 5) exp + denominator segment-sum
    edge_exp_sum<<<(E * NHEADS + 255) / 256, 256, 0, stream>>>(dst, raw, attn_sum, maxscr, E);

    // 6) normalized weighted scatter-add
    aggregate<<<4096, 256, 0, stream>>>(src, dst, h_ws, raw, attn_sum, out, E * DMODEL);
}
